// Encoder_32452772888844
// MI455X (gfx1250) — matrix operand order repaired and checked by reading, not compiled, not run on hardware
//
#include <hip/hip_runtime.h>
#include <hip/hip_bf16.h>
#include <math.h>

// ---------------------------------------------------------------------------
// Encoder: 64 independent 2-layer LSTMs (T=10, IN=H=256) + per-group Linear +
// shared heads. HBM-bound (~285MB fp32 weights); all GEMM-shaped math goes
// through v_wmma_f32_16x16x32_bf16 with T padded to one M=16 tile.
// Cache policy: single-use Wih streams are non-temporal; timestep-reused Whh
// (134MB, 10x reuse) stays L2-resident (192MB L2).
// ---------------------------------------------------------------------------

typedef __attribute__((ext_vector_type(16))) __bf16 bf16x16;
typedef __attribute__((ext_vector_type(8)))  __bf16 bf16x8;
typedef __attribute__((ext_vector_type(8)))  float  f32x8;
typedef __attribute__((ext_vector_type(4)))  float  f32x4;   // clang ext vector: OK for nontemporal builtin

#define NG     64
#define NT     10
#define NH     256
#define FOURH  1024

// out layout (floats): gamma[64][10][1] | beta[64][10][256] | delta[64][10][256]
//                      | hN[64][2][256] | cN[64][2][256]
#define OFF_GAMMA 0
#define OFF_BETA  640
#define OFF_DELTA 164480
#define OFF_HN    328320
#define OFF_CN    361088

// ws layout (floats): xg[64][16][1024] | lin1[64][16][256]
#define WS_XG_STRIDE   (16 * FOURH)
#define WS_LIN1_BASE   ((size_t)NG * WS_XG_STRIDE)
#define WS_LIN1_STRIDE (16 * NH)

// A fragment (16x32 bf16, M x K) from LDS row-major [16][256] bf16.
// Lane l: M = l&15; lanes 0-15 hold K = kt*32 + 0..15,
//                   lanes 16-31 hold K = kt*32 + 16..31.
__device__ __forceinline__ bf16x16 load_a_frag(const __bf16* abuf, int lane, int kt) {
    int m  = lane & 15;
    int kb = (lane >> 4) << 4;                 // 0 or 16
    const __bf16* p = abuf + m * NH + kt * 32 + kb;
    bf16x8 lo = *(const bf16x8*)p;             // K = kb .. kb+7
    bf16x8 hi = *(const bf16x8*)(p + 8);       // K = kb+8 .. kb+15
    return __builtin_shufflevector(lo, hi, 0,1,2,3,4,5,6,7,8,9,10,11,12,13,14,15);
}

// B fragment (32x16 bf16, K x N) with B = W^T, W row-major [Ntot][256] fp32.
// Lane l: N = l&15; lanes 0-15 hold K = kt*32 + 0..15, lanes 16-31 K = +16..31.
// => each lane reads 16 contiguous fp32 of row (ntile*16 + n) and converts.
template <bool NTHINT>
__device__ __forceinline__ bf16x16 load_b_frag(const float* W, int ntile, int kt, int lane) {
    int n  = lane & 15;
    int kb = (lane >> 4) << 4;                 // 0 or 16
    const f32x4* p = (const f32x4*)(W + (size_t)(ntile * 16 + n) * NH + kt * 32 + kb);
    f32x4 w0, w1, w2, w3;
    if (NTHINT) {
        w0 = __builtin_nontemporal_load(p + 0);
        w1 = __builtin_nontemporal_load(p + 1);
        w2 = __builtin_nontemporal_load(p + 2);
        w3 = __builtin_nontemporal_load(p + 3);
    } else {
        w0 = p[0]; w1 = p[1]; w2 = p[2]; w3 = p[3];
    }
    bf16x16 b;
    b[0]  = (__bf16)w0[0]; b[1]  = (__bf16)w0[1]; b[2]  = (__bf16)w0[2]; b[3]  = (__bf16)w0[3];
    b[4]  = (__bf16)w1[0]; b[5]  = (__bf16)w1[1]; b[6]  = (__bf16)w1[2]; b[7]  = (__bf16)w1[3];
    b[8]  = (__bf16)w2[0]; b[9]  = (__bf16)w2[1]; b[10] = (__bf16)w2[2]; b[11] = (__bf16)w2[3];
    b[12] = (__bf16)w3[0]; b[13] = (__bf16)w3[1]; b[14] = (__bf16)w3[2]; b[15] = (__bf16)w3[3];
    return b;
}

// One 16x16 output tile of  A[16x256] * W^T  (K = 256 -> 8 WMMA k-steps).
template <bool NTHINT>
__device__ __forceinline__ f32x8 gemm_tile(const __bf16* abuf, const float* W,
                                           int ntile, int lane) {
    f32x8 c = {};
#pragma unroll
    for (int kt = 0; kt < 8; ++kt) {
        bf16x16 a = load_a_frag(abuf, lane, kt);
        bf16x16 b = load_b_frag<NTHINT>(W, ntile, kt, lane);
        c = __builtin_amdgcn_wmma_f32_16x16x32_bf16(false, a, false, b,
                                                    (short)0, c, false, false);
    }
    return c;
}

__device__ __forceinline__ float sigmoidf_(float x) { return 1.f / (1.f + __expf(-x)); }

__global__ __launch_bounds__(256)
void encoder_kernel(const float* __restrict__ data,
                    const float* __restrict__ Wih0, const float* __restrict__ Whh0,
                    const float* __restrict__ bih0, const float* __restrict__ bhh0,
                    const float* __restrict__ Wih1, const float* __restrict__ Whh1,
                    const float* __restrict__ bih1, const float* __restrict__ bhh1,
                    const float* __restrict__ Wlin, const float* __restrict__ blin,
                    const float* __restrict__ W1,   const float* __restrict__ b1,
                    const float* __restrict__ W2,   const float* __restrict__ b2,
                    const float* __restrict__ Wd,   const float* __restrict__ bd,
                    float* __restrict__ out, float* __restrict__ ws)
{
    __shared__ __bf16 abuf [16 * NH];   // activation A-matrix (bf16, row-major)
    __shared__ __bf16 abuf2[16 * NH];   // fc result as next A-matrix
    __shared__ float  hvec[NH];
    __shared__ float  cvec[NH];
    __shared__ float  lbuf[16 * NH];    // lin1 (for gamma)

    const int g    = blockIdx.x;
    const int tid  = threadIdx.x;
    const int lane = tid & 31;
    const int wave = tid >> 5;
    const int moff = (lane >> 4) << 3;  // D rows: m = v + moff
    const int ncol = lane & 15;

    float* xg   = ws + (size_t)g * WS_XG_STRIDE;
    float* l1ws = ws + WS_LIN1_BASE + (size_t)g * WS_LIN1_STRIDE;

    // ---- stage x = data[:, g, :] into abuf (pad rows 10..15 with zero) ----
    for (int idx = tid; idx < 16 * NH; idx += 256) {
        int t = idx >> 8, k = idx & 255;
        float v = 0.f;
        if (t < NT)
            v = __builtin_nontemporal_load(data + ((size_t)t * NG + g) * NH + k);
        abuf[idx] = (__bf16)v;
    }
    __syncthreads();

    // =========================== two LSTM layers ===========================
    for (int layer = 0; layer < 2; ++layer) {
        const float* Wih = layer ? Wih1 : Wih0;
        const float* Whh = layer ? Whh1 : Whh0;
        const float* bih = layer ? bih1 : bih0;
        const float* bhh = layer ? bhh1 : bhh0;

        // xg[t][4H] = x_t . Wih^T + (bih + bhh)  (WMMA; 64 N-tiles over 8 waves)
        // Wih is single-use -> non-temporal to keep Whh L2-resident.
        const float* Wg = Wih + (size_t)g * FOURH * NH;
        for (int nt = wave; nt < FOURH / 16; nt += 8) {
            f32x8 c = gemm_tile<true>(abuf, Wg, nt, lane);
            int col = nt * 16 + ncol;
            float bias = bih[g * FOURH + col] + bhh[g * FOURH + col];
#pragma unroll
            for (int v = 0; v < 8; ++v) {
                int m = v + moff;
                if (m < NT) xg[(size_t)m * FOURH + col] = c[v] + bias;
            }
        }
        hvec[tid] = 0.f;
        cvec[tid] = 0.f;
        __threadfence();     // make xg (global ws) visible within the block
        __syncthreads();

        // ---- recurrence: thread tid owns hidden index tid, all 4 gates ----
        // Whh streamed with default (RT) policy: re-read every timestep, L2-resident.
        const float4* w0 = (const float4*)(Whh + ((size_t)g * FOURH +   0 + tid) * NH);
        const float4* w1 = (const float4*)(Whh + ((size_t)g * FOURH + 256 + tid) * NH);
        const float4* w2 = (const float4*)(Whh + ((size_t)g * FOURH + 512 + tid) * NH);
        const float4* w3 = (const float4*)(Whh + ((size_t)g * FOURH + 768 + tid) * NH);

        for (int t = 0; t < NT; ++t) {
            const float* row = xg + (size_t)t * FOURH;
            float ai = row[tid], af = row[256 + tid];
            float ag = row[512 + tid], ao = row[768 + tid];
            const float4* h4 = (const float4*)hvec;
#pragma unroll 4
            for (int kk = 0; kk < 64; ++kk) {
                float4 hh = h4[kk];
                float4 x0 = w0[kk], x1 = w1[kk], x2 = w2[kk], x3 = w3[kk];
                ai += x0.x*hh.x + x0.y*hh.y + x0.z*hh.z + x0.w*hh.w;
                af += x1.x*hh.x + x1.y*hh.y + x1.z*hh.z + x1.w*hh.w;
                ag += x2.x*hh.x + x2.y*hh.y + x2.z*hh.z + x2.w*hh.w;
                ao += x3.x*hh.x + x3.y*hh.y + x3.z*hh.z + x3.w*hh.w;
            }
            float cn = sigmoidf_(af) * cvec[tid] + sigmoidf_(ai) * tanhf(ag);
            float hn = sigmoidf_(ao) * tanhf(cn);
            __syncthreads();               // everyone done reading hvec
            hvec[tid] = hn;
            cvec[tid] = cn;
            abuf[t * NH + tid] = (__bf16)hn;   // becomes next layer's A row
            __syncthreads();
        }

        // final states
        out[OFF_HN + (size_t)g * 512 + layer * NH + tid] = hvec[tid];
        out[OFF_CN + (size_t)g * 512 + layer * NH + tid] = cvec[tid];
        __syncthreads();
    }

    // ---- fc = h2 . Wlin^T + blin  ->  abuf2 (bf16 A-matrix) ----
    const float* Wl = Wlin + (size_t)g * NH * NH;
    for (int nt = wave; nt < 16; nt += 8) {
        f32x8 c = gemm_tile<false>(abuf, Wl, nt, lane);
        int col  = nt * 16 + ncol;
        float bb = blin[g * NH + col];
#pragma unroll
        for (int v = 0; v < 8; ++v)
            abuf2[(v + moff) * NH + col] = (__bf16)(c[v] + bb);
    }
    __syncthreads();

    // ---- lin1 = fc . W1^T + b1  ->  lbuf + ws (for cross-group softmax) ----
    for (int nt = wave; nt < 16; nt += 8) {
        f32x8 c = gemm_tile<false>(abuf2, W1, nt, lane);
        int col  = nt * 16 + ncol;
        float bb = b1[col];
#pragma unroll
        for (int v = 0; v < 8; ++v) {
            int m = v + moff;
            float val = c[v] + bb;
            lbuf[m * NH + col] = val;
            if (m < NT) l1ws[(size_t)m * NH + col] = val;
        }
    }

    // ---- beta = softplus(fc . W2^T + b2)  ->  out ----
    for (int nt = wave; nt < 16; nt += 8) {
        f32x8 c = gemm_tile<false>(abuf2, W2, nt, lane);
        int col  = nt * 16 + ncol;
        float bb = b2[col];
#pragma unroll
        for (int v = 0; v < 8; ++v) {
            int m = v + moff;
            if (m < NT) {
                float x  = c[v] + bb;
                float sp = (x > 20.f) ? x : log1pf(__expf(x));
                out[OFF_BETA + (size_t)g * (NT * NH) + m * NH + col] = sp;
            }
        }
    }
    __syncthreads();

    // ---- gamma = lin1 . Wd^T + bd  (10 tiny dots) ----
    if (tid < NT) {
        float a = bd[0];
        const float* lr = lbuf + tid * NH;
        for (int k = 0; k < NH; ++k) a += lr[k] * Wd[k];
        out[OFF_GAMMA + (size_t)g * NT + tid] = a;
    }
}

// delta = softmax over the group axis (dim 0) of lin1
__global__ __launch_bounds__(256)
void delta_softmax_kernel(const float* __restrict__ lin1, float* __restrict__ out)
{
    int t = blockIdx.x;        // 0..9
    int h = threadIdx.x;       // 0..255
    const float* p = lin1 + (size_t)t * NH + h;

    float m = -3.4e38f;
    for (int g = 0; g < NG; ++g) m = fmaxf(m, p[(size_t)g * WS_LIN1_STRIDE]);
    float s = 0.f;
    for (int g = 0; g < NG; ++g) s += __expf(p[(size_t)g * WS_LIN1_STRIDE] - m);
    float inv = 1.f / s;
    float* dst = out + OFF_DELTA + (size_t)t * NH + h;
    for (int g = 0; g < NG; ++g)
        dst[(size_t)g * (NT * NH)] = __expf(p[(size_t)g * WS_LIN1_STRIDE] - m) * inv;
}

extern "C" void kernel_launch(void* const* d_in, const int* in_sizes, int n_in,
                              void* d_out, int out_size, void* d_ws, size_t ws_size,
                              hipStream_t stream) {
    const float* data = (const float*)d_in[0];
    const float* Wih0 = (const float*)d_in[1];
    const float* Whh0 = (const float*)d_in[2];
    const float* bih0 = (const float*)d_in[3];
    const float* bhh0 = (const float*)d_in[4];
    const float* Wih1 = (const float*)d_in[5];
    const float* Whh1 = (const float*)d_in[6];
    const float* bih1 = (const float*)d_in[7];
    const float* bhh1 = (const float*)d_in[8];
    const float* Wlin = (const float*)d_in[9];
    const float* blin = (const float*)d_in[10];
    const float* W1   = (const float*)d_in[11];
    const float* b1   = (const float*)d_in[12];
    const float* W2   = (const float*)d_in[13];
    const float* b2   = (const float*)d_in[14];
    const float* Wd   = (const float*)d_in[15];
    const float* bd   = (const float*)d_in[16];
    float* out = (float*)d_out;
    float* ws  = (float*)d_ws;

    encoder_kernel<<<NG, 256, 0, stream>>>(data, Wih0, Whh0, bih0, bhh0,
                                           Wih1, Whh1, bih1, bhh1,
                                           Wlin, blin, W1, b1, W2, b2, Wd, bd,
                                           out, ws);
    delta_softmax_kernel<<<NT, 256, 0, stream>>>(ws + WS_LIN1_BASE, out);
}
